// PiFoldAttn_15496242004734
// MI455X (gfx1250) — matrix operand order repaired and checked by reading, not compiled, not run on hardware
//
#include <hip/hip_runtime.h>
#include <math.h>

// ---------------- problem constants (from reference) ----------------
static constexpr int    N_  = 20000;
static constexpr long   E_  = 640000;      // N*K
static constexpr int    H_  = 128;
static constexpr int    NV_ = 128;
static constexpr int    NE_ = 128;
static constexpr int    NH_ = 4;
static constexpr int    CATW = 2 * NV_ + NE_;   // 384
// padded LDS row strides (bf16 elements): stride/2 dwords % 64 banks == 4
static constexpr int    CATP = 392;             // 784 B = 196 dw, 196%64=4
static constexpr int    XP   = 136;             // 272 B =  68 dw,  68%64=4

typedef __attribute__((ext_vector_type(16))) __bf16 v16bf;
typedef __attribute__((ext_vector_type(8)))  __bf16 v8bf;
typedef __attribute__((ext_vector_type(4)))  __bf16 v4bf;
typedef __attribute__((ext_vector_type(8)))  float  v8f;
typedef __attribute__((ext_vector_type(4)))  float  v4f;

__device__ inline v8f wmma_bf16(v16bf a, v16bf b, v8f c) {
    return __builtin_amdgcn_wmma_f32_16x16x32_bf16(
        false, a, false, b, (short)0, c, false, false);
}

// A-fragment (16x32 bf16, MxK), row-major source with row stride `stride`.
// Contiguous per half: elems 0-15 -> K = kbase+16*half+0..15
// => two 16B LDS loads per lane.
__device__ inline v16bf load_A(const __bf16* base, int stride, int kbase, int lane) {
    const int half8 = (lane >> 4) * 16;
    const int m     = lane & 15;
    const __bf16* p = base + m * stride + kbase + half8;
    v8bf lo = *(const v8bf*)(p);
    v8bf hi = *(const v8bf*)(p + 8);
    return __builtin_shufflevector(lo, hi, 0, 1, 2, 3, 4, 5, 6, 7,
                                   8, 9, 10, 11, 12, 13, 14, 15);
}

// B-fragment (32x16 bf16, KxN) with B[k][n] = W[n][k] (W row-major [N][K]):
// lane n reads 16 contiguous bf16 (32B) of weight row n.
__device__ inline v16bf load_B(const __bf16* W, int K, int n0, int kbase, int lane) {
    const int n = n0 + (lane & 15);
    const int k = kbase + 16 * (lane >> 4);
    return *(const v16bf*)(W + (long)n * K + k);
}

__device__ inline void atomicMaxF(float* addr, float val) {
    unsigned int* ua = (unsigned int*)addr;
    unsigned int old = *ua;
    while (true) {
        float f = __uint_as_float(old);
        if (f >= val) break;
        unsigned int assumed = old;
        old = atomicCAS(ua, assumed, __float_as_uint(val));
        if (old == assumed) break;
    }
}

// ---------------- kernel 0a: f32 -> bf16 weight conversion ----------------
__global__ void cvt_bf16_kernel(const float* __restrict__ s, __bf16* __restrict__ d, int n) {
    int i = blockIdx.x * 256 + threadIdx.x;
    if (i < n) d[i] = (__bf16)s[i];
}

// B3 (4x128) padded to (16x128) so the logits GEMM is one 16x16 WMMA tile.
__global__ void cvt_b3_kernel(const float* __restrict__ s, __bf16* __restrict__ d) {
    int i = blockIdx.x * 256 + threadIdx.x;
    if (i < 16 * 128) {
        int r = i >> 7;
        d[i] = (r < NH_) ? (__bf16)s[i] : (__bf16)0.0f;
    }
}

// ---------------- kernel 0b: init softmax stats + agg ----------------
__global__ void init_kernel(float* __restrict__ m, float* __restrict__ dn, float* __restrict__ ag) {
    int i = blockIdx.x * 256 + threadIdx.x;
    if (i < N_ * NH_) { m[i] = -INFINITY; dn[i] = 0.0f; }
    if (i < N_ * H_)  ag[i] = 0.0f;
}

// ---------------- kernel 1: per-edge WMMA pipeline ----------------
// 1 wave == one 16-edge tile; 4 waves / block; dynamic LDS, per-wave regions.
__global__ __launch_bounds__(128) void edge_kernel(
    const float* __restrict__ hV, const float* __restrict__ hE,
    const long long* __restrict__ eidx,
    const __bf16* __restrict__ B1, const float* __restrict__ b1,
    const __bf16* __restrict__ B2, const float* __restrict__ b2,
    const __bf16* __restrict__ B3, const float* __restrict__ b3,
    const __bf16* __restrict__ Wv, const float* __restrict__ bv,
    float* __restrict__ logits, __bf16* __restrict__ Vout, float* __restrict__ mmax)
{
    extern __shared__ __bf16 dynlds[];
    const int wave = threadIdx.x >> 5;
    const int lane = threadIdx.x & 31;
    const long tile = (long)blockIdx.x * 4 + wave;
    const long e0   = tile * 16;

    __bf16* cat = dynlds + wave * (16 * CATP + 16 * XP);  // 16 x 384 (stride CATP)
    __bf16* x1  = cat + 16 * CATP;                        // 16 x 128 (stride XP)

    // ---- build cat = [hV[src] | hE | hV[dst]] in bf16, float4-vectorized
    for (int t = lane; t < 16 * 96; t += 32) {
        const int r = t / 96, c4 = t - r * 96;
        const long e = e0 + r;
        v4f f;
        if (c4 < 32)      f = *(const v4f*)(hV + eidx[e] * NV_ + c4 * 4);
        else if (c4 < 64) f = *(const v4f*)(hE + e * NE_ + (c4 - 32) * 4);
        else              f = *(const v4f*)(hV + eidx[E_ + e] * NV_ + (c4 - 64) * 4);
        *(v4bf*)(cat + r * CATP + c4 * 4) = __builtin_convertvector(f, v4bf);
    }
    const int n_lo  = lane & 15;
    const int mrow0 = (lane >> 4) * 8;

    // ---- GEMM1: x1 = relu(cat @ B1^T + b1)   [16x384]x[384x128]
    {
        v8f acc[8] = {};
#pragma unroll
        for (int kt = 0; kt < CATW / 32; ++kt) {
            v16bf a = load_A(cat, CATP, kt * 32, lane);
#pragma unroll
            for (int nt = 0; nt < 8; ++nt)
                acc[nt] = wmma_bf16(a, load_B(B1, CATW, nt * 16, kt * 32, lane), acc[nt]);
        }
#pragma unroll
        for (int nt = 0; nt < 8; ++nt) {
            const int n = nt * 16 + n_lo;
            const float bias = b1[n];
#pragma unroll
            for (int r = 0; r < 8; ++r) {
                float v = acc[nt][r] + bias;
                x1[(mrow0 + r) * XP + n] = (__bf16)(v > 0.0f ? v : 0.0f);
            }
        }
    }

    // ---- V = gelu(hE @ Wv^T + bv)  (A = cat columns 128..255), store bf16
    {
        v8f acc[8] = {};
#pragma unroll
        for (int kt = 0; kt < 4; ++kt) {
            v16bf a = load_A(cat + NV_, CATP, kt * 32, lane);
#pragma unroll
            for (int nt = 0; nt < 8; ++nt)
                acc[nt] = wmma_bf16(a, load_B(Wv, NE_, nt * 16, kt * 32, lane), acc[nt]);
        }
#pragma unroll
        for (int nt = 0; nt < 8; ++nt) {
            const int n = nt * 16 + n_lo;
            const float bias = bv[n];
#pragma unroll
            for (int r = 0; r < 8; ++r) {
                float v = acc[nt][r] + bias;
                float g = 0.5f * v * (1.0f + erff(v * 0.70710678118654752f));
                Vout[(e0 + mrow0 + r) * H_ + n] = (__bf16)g;
            }
        }
    }

    // ---- GEMM2: x2 = relu(x1 @ B2^T + b2); x2 overlays the cat region
    __bf16* x2 = cat;   // stride XP, 16*136 elems < 16*CATP: safe overlay
    {
        v8f acc[8] = {};
#pragma unroll
        for (int kt = 0; kt < 4; ++kt) {
            v16bf a = load_A(x1, XP, kt * 32, lane);
#pragma unroll
            for (int nt = 0; nt < 8; ++nt)
                acc[nt] = wmma_bf16(a, load_B(B2, H_, nt * 16, kt * 32, lane), acc[nt]);
        }
#pragma unroll
        for (int nt = 0; nt < 8; ++nt) {
            const int n = nt * 16 + n_lo;
            const float bias = b2[n];
#pragma unroll
            for (int r = 0; r < 8; ++r) {
                float v = acc[nt][r] + bias;
                x2[(mrow0 + r) * XP + n] = (__bf16)(v > 0.0f ? v : 0.0f);
            }
        }
    }

    // ---- GEMM3: logits = (x2 @ B3pad^T + b3) / sqrt(D); atomic segment-max
    {
        v8f acc = {};
#pragma unroll
        for (int kt = 0; kt < 4; ++kt)
            acc = wmma_bf16(load_A(x2, XP, kt * 32, lane),
                            load_B(B3, H_, 0, kt * 32, lane), acc);
        const float scale = 0.17677669529663689f;   // 1/sqrt(32)
        if (n_lo < NH_) {
            const float bias = b3[n_lo];
#pragma unroll
            for (int r = 0; r < 8; ++r) {
                const long e = e0 + mrow0 + r;
                float lg = (acc[r] + bias) * scale;
                logits[e * NH_ + n_lo] = lg;
                atomicMaxF(&mmax[eidx[e] * NH_ + n_lo], lg);
            }
        }
    }
}

// ---------------- kernel 2: softmax denominator ----------------
__global__ void denom_kernel(const float* __restrict__ logits, const long long* __restrict__ eidx,
                             const float* __restrict__ mmax, float* __restrict__ denom)
{
    long i = (long)blockIdx.x * 256 + threadIdx.x;
    if (i >= E_ * NH_) return;
    const long e = i >> 2;
    const int  h = (int)(i & 3);
    const long s = eidx[e];
    atomicAdd(&denom[s * NH_ + h], expf(logits[i] - mmax[s * NH_ + h]));
}

// ---------------- kernel 3: weighted scatter aggregation ----------------
__global__ void agg_kernel(const float* __restrict__ logits, const long long* __restrict__ eidx,
                           const float* __restrict__ mmax, const float* __restrict__ denom,
                           const __bf16* __restrict__ Vbf, float* __restrict__ agg)
{
    long i = (long)blockIdx.x * 256 + threadIdx.x;
    if (i >= E_ * (long)H_) return;
    const long e = i >> 7;
    const int  c = (int)(i & 127);
    const int  h = c >> 5;
    const long s = eidx[e];
    const float at = expf(logits[e * NH_ + h] - mmax[s * NH_ + h]) / denom[s * NH_ + h];
    atomicAdd(&agg[s * H_ + c], at * (float)Vbf[i]);
}

// ---------------- kernel 4: node-level gated output ----------------
__global__ __launch_bounds__(128) void node_kernel(
    const float* __restrict__ agg, const float* __restrict__ hV,
    const __bf16* __restrict__ Wo, const __bf16* __restrict__ Gw,
    const float* __restrict__ gb, float* __restrict__ out)
{
    __shared__ __bf16 abf[4][16 * XP];
    __shared__ float  gf[4][16 * H_];
    const int wave = threadIdx.x >> 5;
    const int lane = threadIdx.x & 31;
    const int tile = blockIdx.x * 4 + wave;
    if (tile >= N_ / 16) return;
    const long r0 = (long)tile * 16;

    __bf16* A = abf[wave];
    float*  G = gf[wave];
    for (int t = lane; t < 16 * 32; t += 32) {
        const int r = t >> 5, c4 = t & 31;
        v4f f = *(const v4f*)(agg + (r0 + r) * H_ + c4 * 4);
        *(v4bf*)(A + r * XP + c4 * 4) = __builtin_convertvector(f, v4bf);
    }

    const int n_lo  = lane & 15;
    const int mrow0 = (lane >> 4) * 8;

    // gate = sigmoid(agg @ gate_w^T + gate_b)
    {
        v8f acc[8] = {};
#pragma unroll
        for (int kt = 0; kt < 4; ++kt) {
            v16bf a = load_A(A, XP, kt * 32, lane);
#pragma unroll
            for (int nt = 0; nt < 8; ++nt)
                acc[nt] = wmma_bf16(a, load_B(Gw, H_, nt * 16, kt * 32, lane), acc[nt]);
        }
#pragma unroll
        for (int nt = 0; nt < 8; ++nt) {
            const int n = nt * 16 + n_lo;
            const float bias = gb[n];
#pragma unroll
            for (int r = 0; r < 8; ++r) {
                float v = acc[nt][r] + bias;
                G[(mrow0 + r) * H_ + n] = 1.0f / (1.0f + expf(-v));
            }
        }
    }
    // out = h_V + (agg @ Wo^T) * gate
    {
        v8f acc[8] = {};
#pragma unroll
        for (int kt = 0; kt < 4; ++kt) {
            v16bf a = load_A(A, XP, kt * 32, lane);
#pragma unroll
            for (int nt = 0; nt < 8; ++nt)
                acc[nt] = wmma_bf16(a, load_B(Wo, H_, nt * 16, kt * 32, lane), acc[nt]);
        }
#pragma unroll
        for (int nt = 0; nt < 8; ++nt) {
            const int n = nt * 16 + n_lo;
#pragma unroll
            for (int r = 0; r < 8; ++r) {
                const long row = r0 + mrow0 + r;
                out[row * H_ + n] = hV[row * H_ + n] + acc[nt][r] * G[(mrow0 + r) * H_ + n];
            }
        }
    }
}

// ---------------- host-side orchestration ----------------
extern "C" void kernel_launch(void* const* d_in, const int* in_sizes, int n_in,
                              void* d_out, int out_size, void* d_ws, size_t ws_size,
                              hipStream_t stream)
{
    const float*     hV   = (const float*)d_in[0];
    const float*     hE   = (const float*)d_in[1];
    const long long* eidx = (const long long*)d_in[2];
    const float* Wv_w = (const float*)d_in[3];  const float* Wv_b = (const float*)d_in[4];
    const float* B1_w = (const float*)d_in[5];  const float* B1_b = (const float*)d_in[6];
    const float* B2_w = (const float*)d_in[7];  const float* B2_b = (const float*)d_in[8];
    const float* B3_w = (const float*)d_in[9];  const float* B3_b = (const float*)d_in[10];
    const float* Wo_w = (const float*)d_in[11];
    const float* Gw_w = (const float*)d_in[12]; const float* Gw_b = (const float*)d_in[13];
    float* out = (float*)d_out;

    // workspace layout (256B-aligned sections)
    char* ws = (char*)d_ws;
    size_t off = 0;
    auto carve = [&](size_t bytes) { char* p = ws + off; off = (off + bytes + 255) & ~size_t(255); return p; };
    __bf16* B1bf = (__bf16*)carve((size_t)H_ * CATW * 2);
    __bf16* B2bf = (__bf16*)carve((size_t)H_ * H_ * 2);
    __bf16* B3bf = (__bf16*)carve((size_t)16 * H_ * 2);
    __bf16* Wvbf = (__bf16*)carve((size_t)H_ * NE_ * 2);
    __bf16* Wobf = (__bf16*)carve((size_t)NV_ * H_ * 2);
    __bf16* Gwbf = (__bf16*)carve((size_t)NV_ * H_ * 2);
    float*  mmax  = (float*)carve((size_t)N_ * NH_ * 4);
    float*  denom = (float*)carve((size_t)N_ * NH_ * 4);
    float*  logit = (float*)carve((size_t)E_ * NH_ * 4);
    float*  agg   = (float*)carve((size_t)N_ * H_ * 4);
    __bf16* Vbf   = (__bf16*)carve((size_t)E_ * H_ * 2);

    // 0) weight conversion + stat init
    cvt_bf16_kernel<<<(H_ * CATW + 255) / 256, 256, 0, stream>>>(B1_w, B1bf, H_ * CATW);
    cvt_bf16_kernel<<<(H_ * H_ + 255) / 256, 256, 0, stream>>>(B2_w, B2bf, H_ * H_);
    cvt_b3_kernel  <<<(16 * H_ + 255) / 256, 256, 0, stream>>>(B3_w, B3bf);
    cvt_bf16_kernel<<<(H_ * NE_ + 255) / 256, 256, 0, stream>>>(Wv_w, Wvbf, H_ * NE_);
    cvt_bf16_kernel<<<(NV_ * H_ + 255) / 256, 256, 0, stream>>>(Wo_w, Wobf, NV_ * H_);
    cvt_bf16_kernel<<<(NV_ * H_ + 255) / 256, 256, 0, stream>>>(Gw_w, Gwbf, NV_ * H_);
    init_kernel<<<(N_ * H_ + 255) / 256, 256, 0, stream>>>(mmax, denom, agg);

    // 1) per-edge MLP + value projection + logits (all WMMA), segment max
    const size_t edge_lds = 4u * (16 * CATP + 16 * XP) * sizeof(__bf16);  // 67584 B
    edge_kernel<<<(int)(E_ / 64), 128, edge_lds, stream>>>(
        hV, hE, eidx, B1bf, B1_b, B2bf, B2_b, B3bf, B3_b, Wvbf, Wv_b,
        logit, Vbf, mmax);

    // 2) softmax denominator
    denom_kernel<<<(int)((E_ * NH_ + 255) / 256), 256, 0, stream>>>(logit, eidx, mmax, denom);

    // 3) weighted scatter-sum
    agg_kernel<<<(int)((E_ * (long)H_ + 255) / 256), 256, 0, stream>>>(
        logit, eidx, mmax, denom, Vbf, agg);

    // 4) node-level gated residual output (WMMA)
    node_kernel<<<(N_ / 16 + 3) / 4, 128, 0, stream>>>(agg, hV, Wobf, Gwbf, Gw_b, out);

    (void)in_sizes; (void)n_in; (void)out_size; (void)ws_size;
}
